// MMoE_Layer_15522011808414
// MI455X (gfx1250) — matrix operand order repaired and checked by reading, not compiled, not run on hardware
//
#include <hip/hip_runtime.h>
#include <hip/hip_bf16.h>

// MMoE fully fused, f16 WMMA (f32 accum) on gfx1250.
// B=262144, D=128, E=8, T=3, H1=64, H2=32, G=64.

#define BATCH   262144
#define NBLOCKS 256          // workgroups
#define ROWS_WG 1024         // rows per workgroup
#define NITER   4            // 8 waves * 32 rows * 4 iters = 1024

typedef __attribute__((ext_vector_type(16))) _Float16 v16h;
typedef __attribute__((ext_vector_type(8)))  _Float16 v8h;
typedef __attribute__((ext_vector_type(8)))  float    v8f;

// ---------------- LDS layout ----------------
// Fragment region (halves): 214 fragments x 512 halves (1KB each, lane-major)
#define FR_E1    0            // 128 frags: ((e*4+n)*4+k)       We1: K=128,N=64
#define FR_E2    (128*512)    //  32 frags: ((e*2+n2)*2+k2)     We2: K=64, N=32
#define FR_G1    (160*512)    //  48 frags: ((t*4+n)*4+k)       Wg1: K=128,N=64
#define FR_G2    (208*512)    //   6 frags: (t*2+k2)            Wg2: K=64, N=16(pad)
#define FR_TOTAL (214*512)    // 109568 halves = 219136 bytes

#define BIAS_F32_OFF (FR_TOTAL/2)  // float-index into the same LDS (54784)
#define B_E1 0                     // 512 floats
#define B_E2 512                   // 256
#define B_G1 768                   // 192
#define B_G2 960                   // 48 (padded to 16 per task)
#define BIAS_BYTES 4096

#define WAVE_OFF_B  (219136 + BIAS_BYTES)     // per-wave staging base (bytes)
#define STG_ROWPAD  72                        // halves per staged row (bank-conflict pad)
#define STG_BYTES   (32*STG_ROWPAD*2)         // 4608 B : 2 m-tiles x 16 rows x 64(+8) cols
#define GATE_ROWPAD 24                        // halves per gate row (3 tasks x 8 experts)
#define GATEL_BYTES (32*GATE_ROWPAD*2)        // 1536 B
#define WAVE_BYTES  (STG_BYTES + GATEL_BYTES) // 6144 B
#define LDS_TOTAL   (WAVE_OFF_B + 8*WAVE_BYTES) // 272384 B (< 320KB)

static_assert(LDS_TOTAL <= 320*1024, "LDS over WGP budget");

static __device__ __attribute__((always_inline)) inline
v8f wmma_f16(v16h a, v16h b, v8f c) {
    return __builtin_amdgcn_wmma_f32_16x16x32_f16(
        /*neg_a=*/false, a, /*neg_b=*/false, b,
        /*c_mod=*/(short)0, c, /*reuse_a=*/false, /*reuse_b=*/false);
}

static __device__ inline float relu(float v) { return v > 0.0f ? v : 0.0f; }

// B fragment: 1KB, lane-major -> two ds_load_b128 per lane
static __device__ __attribute__((always_inline)) inline
v16h ldfrag(const _Float16* sh, int off, int lane) {
    return *(const v16h*)(sh + off + lane * 16);
}

// A fragment from staged row-major f16 tile (row stride STG_ROWPAD).
// p points at rowStart + k2*32; lane pattern: halves0..7 = K kb..kb+7,
// halves8..15 = K kb+16..kb+23, kb = 8*(lane>=16).
static __device__ __attribute__((always_inline)) inline
v16h ldA2(const _Float16* p, int hl) {
    const _Float16* q = p + hl * 8;
    v8h lo = *(const v8h*)(q);
    v8h hi = *(const v8h*)(q + 16);
    v16h r;
    #pragma unroll
    for (int i = 0; i < 8; ++i) { r[i] = lo[i]; r[i + 8] = hi[i]; }
    return r;
}

// Stage-1 (K=128 GEMM) for one group of 2 N-tiles, 4 accumulator chains,
// 4-slot B-fragment pipeline with prefetch depth 2: the fragment consumed at
// step s was loaded at step s-2 (LDS latency hidden), and the slot written at
// step s was last read at step s-2 (WAR hazard distance >= 4 WMMAs).
static __device__ __attribute__((always_inline)) inline
void stage1_group(const _Float16* sh, _Float16* stg, const float* shf,
                  int biasBase, int fragBase, const v16h (&xa)[2][4],
                  int lane, int hl, int lx, int ng)
{
    v8f acc[2][2];
    acc[0][0] = (v8f){}; acc[0][1] = (v8f){};
    acc[1][0] = (v8f){}; acc[1][1] = (v8f){};
    // fragment sequence: s = k*2 + np, n = ng*2+np -> frag ((n)*4 + k)
    v16h bs[4];
    bs[0] = ldfrag(sh, fragBase + ((ng * 2 + 0) * 4 + 0) * 512, lane);
    bs[1] = ldfrag(sh, fragBase + ((ng * 2 + 1) * 4 + 0) * 512, lane);
    #pragma unroll
    for (int s = 0; s < 8; ++s) {
        if (s + 2 < 8) {
            const int kp = (s + 2) >> 1, npp = (s + 2) & 1;
            bs[(s + 2) & 3] =
                ldfrag(sh, fragBase + ((ng * 2 + npp) * 4 + kp) * 512, lane);
        }
        const int k = s >> 1, np = s & 1;
        acc[np][0] = wmma_f16(xa[0][k], bs[s & 3], acc[np][0]);
        acc[np][1] = wmma_f16(xa[1][k], bs[s & 3], acc[np][1]);
    }
    #pragma unroll
    for (int np = 0; np < 2; ++np) {
        const int n = ng * 2 + np;
        const float bv = shf[biasBase + n * 16 + lx];
        #pragma unroll
        for (int v = 0; v < 8; ++v) {
            const int rr = v + hl * 8;
            stg[rr * STG_ROWPAD + n * 16 + lx]        = (_Float16)relu(acc[np][0][v] + bv);
            stg[(rr + 16) * STG_ROWPAD + n * 16 + lx] = (_Float16)relu(acc[np][1][v] + bv);
        }
    }
}

__global__ __launch_bounds__(256)
void mmoe_kernel(const float* __restrict__ x,
                 const float* __restrict__ We1, const float* __restrict__ be1,
                 const float* __restrict__ We2, const float* __restrict__ be2,
                 const float* __restrict__ Wg1, const float* __restrict__ bg1,
                 const float* __restrict__ Wg2, const float* __restrict__ bg2,
                 float* __restrict__ out)
{
    extern __shared__ _Float16 sh[];
    float* shf = (float*)sh;

    const int tid  = threadIdx.x;
    const int lane = tid & 31;
    const int wave = tid >> 5;
    const int hl   = lane >> 4;   // half of wave (0: lanes 0-15, 1: lanes 16-31)
    const int lx   = lane & 15;

    // ---- Setup: shuffle all weights into WMMA B-fragment layout (f32 -> f16) ----
    // B-frag element (lane l, half h): K = 8*(l>=16) + (h&7) + 16*(h>=8), N = l&15 (per ISA 05_wmma).
    for (int idx = tid; idx < FR_TOTAL; idx += 256) {
        const int f  = idx >> 9;
        const int r  = idx & 511;
        const int l  = r >> 4;
        const int h  = r & 15;
        const int kl = ((l >> 4) << 3) + (h & 7) + ((h >> 3) << 4);   // K-local in [0,32)
        const int nn = l & 15;
        float v;
        if (f < 128) {                        // We1 [e][128][64]
            const int e = f >> 4, n = (f >> 2) & 3, k = f & 3;
            v = We1[(e * 128 + k * 32 + kl) * 64 + n * 16 + nn];
        } else if (f < 160) {                 // We2 [e][64][32]
            const int g = f - 128, e = g >> 2, n2 = (g >> 1) & 1, k2 = g & 1;
            v = We2[(e * 64 + k2 * 32 + kl) * 32 + n2 * 16 + nn];
        } else if (f < 208) {                 // Wg1 [t][128][64]
            const int g = f - 160, t = g >> 4, n = (g >> 2) & 3, k = g & 3;
            v = Wg1[(t * 128 + k * 32 + kl) * 64 + n * 16 + nn];
        } else {                              // Wg2 [t][64][8], N padded to 16
            const int g = f - 208, t = g >> 1, k2 = g & 1;
            v = (nn < 8) ? Wg2[(t * 64 + k2 * 32 + kl) * 8 + nn] : 0.0f;
        }
        sh[idx] = (_Float16)v;
    }
    for (int idx = tid; idx < 1008; idx += 256) {
        float v;
        if (idx < 512)      v = be1[idx];
        else if (idx < 768) v = be2[idx - 512];
        else if (idx < 960) v = bg1[idx - 768];
        else { const int i = idx - 960, t = i >> 4, c = i & 15;
               v = (c < 8) ? bg2[t * 8 + c] : 0.0f; }
        shf[BIAS_F32_OFF + idx] = v;
    }
    __syncthreads();

    _Float16* stg = sh + WAVE_OFF_B / 2 + wave * (WAVE_BYTES / 2); // 32 x 72 staged tile
    _Float16* gtl = stg + STG_BYTES / 2;                           // 32 x 24 f16 gates

    const int rowWG = blockIdx.x * ROWS_WG;

    for (int it = 0; it < NITER; ++it) {
        const int rowBase = rowWG + it * 256 + wave * 32;

        // ---- Load x tile (32 rows x 128) as resident A-fragments (f32 -> f16) ----
        v16h xa[2][4];
        #pragma unroll
        for (int m = 0; m < 2; ++m) {
            const float* xr0 = x + (size_t)(rowBase + m * 16 + lx) * 128 + hl * 8;
            #pragma unroll
            for (int ks = 0; ks < 4; ++ks) {
                const float* p = xr0 + ks * 32;
                const float4 f0 = *(const float4*)(p);
                const float4 f1 = *(const float4*)(p + 4);
                const float4 f2 = *(const float4*)(p + 16);
                const float4 f3 = *(const float4*)(p + 20);
                v16h a;
                a[0]=(_Float16)f0.x; a[1]=(_Float16)f0.y; a[2]=(_Float16)f0.z; a[3]=(_Float16)f0.w;
                a[4]=(_Float16)f1.x; a[5]=(_Float16)f1.y; a[6]=(_Float16)f1.z; a[7]=(_Float16)f1.w;
                a[8]=(_Float16)f2.x; a[9]=(_Float16)f2.y; a[10]=(_Float16)f2.z; a[11]=(_Float16)f2.w;
                a[12]=(_Float16)f3.x; a[13]=(_Float16)f3.y; a[14]=(_Float16)f3.z; a[15]=(_Float16)f3.w;
                xa[m][ks] = a;
            }
        }

        // ================= Gate path: softmax(relu(x@Wg1+bg1)@Wg2+bg2) =================
        #pragma unroll 1
        for (int t = 0; t < 3; ++t) {
            // stage 1: g = relu(x @ Wg1[t] + bg1[t]) -> staged f16 tile
            stage1_group(sh, stg, shf, BIAS_F32_OFF + B_G1 + t * 64,
                         FR_G1 + t * 16 * 512, xa, lane, hl, lx, 0);
            stage1_group(sh, stg, shf, BIAS_F32_OFF + B_G1 + t * 64,
                         FR_G1 + t * 16 * 512, xa, lane, hl, lx, 1);
            // stage 2: logits = g @ Wg2[t] + bg2[t]  (N padded to 16)
            v8f lg0 = (v8f){}, lg1 = (v8f){};
            #pragma unroll
            for (int k2 = 0; k2 < 2; ++k2) {
                const v16h bf = ldfrag(sh, FR_G2 + (t * 2 + k2) * 512, lane);
                const v16h a20 = ldA2(stg + (0 * 16 + lx) * STG_ROWPAD + k2 * 32, hl);
                const v16h a21 = ldA2(stg + (1 * 16 + lx) * STG_ROWPAD + k2 * 32, hl);
                lg0 = wmma_f16(a20, bf, lg0);
                lg1 = wmma_f16(a21, bf, lg1);
            }
            // softmax over E=8 (lanes lx 0..7 within each wave-half), per row
            const float bv2 = shf[BIAS_F32_OFF + B_G2 + t * 16 + lx];
            #pragma unroll
            for (int m = 0; m < 2; ++m) {
                #pragma unroll
                for (int v = 0; v < 8; ++v) {
                    float z = ((m == 0) ? lg0[v] : lg1[v]) + bv2;
                    float zm = z;
                    zm = fmaxf(zm, __shfl_xor(zm, 1));
                    zm = fmaxf(zm, __shfl_xor(zm, 2));
                    zm = fmaxf(zm, __shfl_xor(zm, 4));
                    const float ez = __expf(z - zm);
                    float s = ez;
                    s += __shfl_xor(s, 1);
                    s += __shfl_xor(s, 2);
                    s += __shfl_xor(s, 4);
                    const float gv = ez / s;
                    if (lx < 8)
                        gtl[(m * 16 + v + hl * 8) * GATE_ROWPAD + t * 8 + lx] = (_Float16)gv;
                }
            }
        }

        // ================= Expert path + softmax-weighted mix =================
        float outAcc[3][2][2][8];
        #pragma unroll
        for (int t = 0; t < 3; ++t)
            #pragma unroll
            for (int m = 0; m < 2; ++m)
                #pragma unroll
                for (int n2 = 0; n2 < 2; ++n2)
                    #pragma unroll
                    for (int v = 0; v < 8; ++v) outAcc[t][m][n2][v] = 0.0f;

        #pragma unroll 1
        for (int e = 0; e < 8; ++e) {
            // stage 1: h1 = relu(x @ We1[e] + be1[e]) -> staged f16 tile
            stage1_group(sh, stg, shf, BIAS_F32_OFF + B_E1 + e * 64,
                         FR_E1 + e * 16 * 512, xa, lane, hl, lx, 0);
            stage1_group(sh, stg, shf, BIAS_F32_OFF + B_E1 + e * 64,
                         FR_E1 + e * 16 * 512, xa, lane, hl, lx, 1);
            // stage 2: eo = relu(h1 @ We2[e] + be2[e])  (4 independent chains)
            v8f eo[2][2];
            #pragma unroll
            for (int n2 = 0; n2 < 2; ++n2) { eo[0][n2] = (v8f){}; eo[1][n2] = (v8f){}; }
            #pragma unroll
            for (int k2 = 0; k2 < 2; ++k2) {
                const v16h a20 = ldA2(stg + (0 * 16 + lx) * STG_ROWPAD + k2 * 32, hl);
                const v16h a21 = ldA2(stg + (1 * 16 + lx) * STG_ROWPAD + k2 * 32, hl);
                #pragma unroll
                for (int n2 = 0; n2 < 2; ++n2) {
                    const v16h bf = ldfrag(sh, FR_E2 + ((e * 2 + n2) * 2 + k2) * 512, lane);
                    eo[0][n2] = wmma_f16(a20, bf, eo[0][n2]);
                    eo[1][n2] = wmma_f16(a21, bf, eo[1][n2]);
                }
            }
            // bias + relu epilogue
            #pragma unroll
            for (int n2 = 0; n2 < 2; ++n2) {
                const float bv = shf[BIAS_F32_OFF + B_E2 + e * 32 + n2 * 16 + lx];
                #pragma unroll
                for (int m = 0; m < 2; ++m)
                    #pragma unroll
                    for (int v = 0; v < 8; ++v)
                        eo[m][n2][v] = relu(eo[m][n2][v] + bv);
            }
            // mix: out[t] += gate[row][t][e] * eo  (gate read = per-half LDS broadcast)
            #pragma unroll
            for (int t = 0; t < 3; ++t)
                #pragma unroll
                for (int m = 0; m < 2; ++m)
                    #pragma unroll
                    for (int v = 0; v < 8; ++v) {
                        const float g =
                            (float)gtl[(m * 16 + v + hl * 8) * GATE_ROWPAD + t * 8 + e];
                        outAcc[t][m][0][v] += g * eo[m][0][v];
                        outAcc[t][m][1][v] += g * eo[m][1][v];
                    }
        }

        // ---- Store: d_out layout [t][b][32] ----
        #pragma unroll
        for (int t = 0; t < 3; ++t)
            #pragma unroll
            for (int m = 0; m < 2; ++m)
                #pragma unroll
                for (int n2 = 0; n2 < 2; ++n2)
                    #pragma unroll
                    for (int v = 0; v < 8; ++v) {
                        const size_t row = (size_t)(rowBase + m * 16 + v + hl * 8);
                        out[(size_t)t * BATCH * 32 + row * 32 + n2 * 16 + lx] =
                            outAcc[t][m][n2][v];
                    }
    }
}

extern "C" void kernel_launch(void* const* d_in, const int* in_sizes, int n_in,
                              void* d_out, int out_size, void* d_ws, size_t ws_size,
                              hipStream_t stream) {
    (void)in_sizes; (void)n_in; (void)d_ws; (void)ws_size; (void)out_size;
    const float* x   = (const float*)d_in[0];
    const float* We1 = (const float*)d_in[1];
    const float* be1 = (const float*)d_in[2];
    const float* We2 = (const float*)d_in[3];
    const float* be2 = (const float*)d_in[4];
    const float* Wg1 = (const float*)d_in[5];
    const float* bg1 = (const float*)d_in[6];
    const float* Wg2 = (const float*)d_in[7];
    const float* bg2 = (const float*)d_in[8];
    float* out = (float*)d_out;

    hipFuncSetAttribute(reinterpret_cast<const void*>(mmoe_kernel),
                        hipFuncAttributeMaxDynamicSharedMemorySize, LDS_TOTAL);
    mmoe_kernel<<<dim3(NBLOCKS), dim3(256), LDS_TOTAL, stream>>>(
        x, We1, be1, We2, be2, Wg1, bg1, Wg2, bg2, out);
}
